// Matcher_11759620457125
// MI455X (gfx1250) — matrix operand order repaired and checked by reading, not compiled, not run on hardware
//
#include <hip/hip_runtime.h>
#include <math.h>

typedef __attribute__((ext_vector_type(16))) _Float16 v16h;
typedef __attribute__((ext_vector_type(8)))  _Float16 v8h;
typedef __attribute__((ext_vector_type(8)))  float    v8f;

#define O_N   3
#define DKEY  128
#define BANK  8192
#define DVAL  512
#define TOPK  64
#define KM_SIG 5.6f
#define LDSTR 136   // padded LDS row stride in halves: 272B = 17*16B

__device__ __forceinline__ unsigned ordered_f32(float f) {
  unsigned u = __float_as_uint(f);
  return (u & 0x80000000u) ? ~u : (u | 0x80000000u);
}

// ---------------------------------------------------------------------------
// Kernel 0: zero the argmax atomic array
// ---------------------------------------------------------------------------
__global__ void k_init(unsigned long long* __restrict__ a, int count) {
  int i = blockIdx.x * 256 + threadIdx.x;
  if (i < count) a[i] = 0ull;
}

// ---------------------------------------------------------------------------
// Kernel 1: p[o,k,n] = (1/sqrt(128)) * keys^T q_in, WMMA f32_16x16x32_f16.
// Block = 256 thr = 8 waves, tile 128(k) x 128(n); wave w owns n-subtile w,
// loops over 8 k-subtiles. Fused epilogue publishes per-row argmax-over-n as
// atomicMax on (orderedFloat(score)<<32 | ~n) via ballot-based lane argmax.
// ---------------------------------------------------------------------------
__global__ __launch_bounds__(256)
void k_scores(const float* __restrict__ keys, const float* __restrict__ qin,
              float* __restrict__ p, unsigned long long* __restrict__ amax64,
              int N) {
  __shared__ _Float16 As[128 * LDSTR] __attribute__((aligned(16))); // [m][d]
  __shared__ _Float16 Bs[128 * LDSTR] __attribute__((aligned(16))); // [n][d]
  const int o     = blockIdx.z;
  const int mbase = blockIdx.y * 128;
  const int nbase = blockIdx.x * 128;
  const int tid   = threadIdx.x;

  for (int idx = tid; idx < DKEY * 128; idx += 256) {
    int d = idx >> 7, ml = idx & 127;
    As[ml * LDSTR + d] = (_Float16)keys[((size_t)o * DKEY + d) * BANK + mbase + ml];
  }
  for (int idx = tid; idx < DKEY * 128; idx += 256) {
    int d = idx >> 7, nl = idx & 127;
    int n = nbase + nl;
    Bs[nl * LDSTR + d] = (n < N) ? (_Float16)qin[(size_t)d * N + n] : (_Float16)0.0f;
  }
  __syncthreads();

  const int wave  = tid >> 5;
  const int lane  = tid & 31;
  const int lhalf = lane & 15;
  const int hi    = lane >> 4;
  const int n     = nbase + wave * 16 + lhalf;   // this lane's output column
  const float scale = 0.088388347648318447f;     // 1/sqrt(128)
  // Per-lane store base: folds o,mbase,8*hi,n once (no per-store 64-bit mul)
  float* const pl = p + ((size_t)o * BANK + mbase + 8 * hi) * N + n;
  const int aidx = o * BANK + mbase + 8 * hi;    // atomic row base (per lane)

  // Hoisted B fragments: one contiguous K-run of 16 per lane (k = 16*hi + s, same order as A)
  v16h bfrag[4];
  #pragma unroll
  for (int c = 0; c < 4; ++c) {
    const _Float16* br = Bs + (wave * 16 + lhalf) * LDSTR + c * 32 + hi * 16;
    v8h b0 = *(const v8h*)(br);
    v8h b1 = *(const v8h*)(br + 8);
    bfrag[c] = __builtin_shufflevector(b0, b1, 0,1,2,3,4,5,6,7,8,9,10,11,12,13,14,15);
  }

  for (int mt = 0; mt < 8; ++mt) {
    const _Float16* ar = As + (mt * 16 + lhalf) * LDSTR;
    v16h af[4];
    #pragma unroll
    for (int c = 0; c < 4; ++c) {
      v8h a0 = *(const v8h*)(ar + c * 32 + hi * 16);
      v8h a1 = *(const v8h*)(ar + c * 32 + hi * 16 + 8);
      af[c] = __builtin_shufflevector(a0, a1, 0,1,2,3,4,5,6,7,8,9,10,11,12,13,14,15);
    }
    v8f cacc = {};
    #pragma unroll
    for (int c = 0; c < 4; ++c)
      cacc = __builtin_amdgcn_wmma_f32_16x16x32_f16(false, af[c], false, bfrag[c],
                                                    (short)0, cacc, false, false);

    // ---- Epilogue: compute everything, then two branch-free-ish regions ----
    float    vals[8];
    unsigned keyhi[8];
    unsigned nwin[8];
    #pragma unroll
    for (int r = 0; r < 8; ++r) {
      float val = cacc[r] * scale;
      vals[r] = val;
      unsigned ordv = (n < N) ? ordered_f32(val) : 0u;
      unsigned m = ordv;
      #pragma unroll
      for (int off = 1; off < 16; off <<= 1)       // stays within 16-lane half
        m = max(m, (unsigned)__shfl_xor((int)m, off));
      unsigned long long bal = __ballot(m == ordv);
      keyhi[r] = m;
      // smallest matching lane -> smallest n (first-occurrence ties)
      nwin[r] = (unsigned)(nbase + wave * 16) +
                (unsigned)__builtin_ctz((unsigned)((bal >> (hi * 16)) & 0xFFFFull));
    }
    if (n < N) {
      #pragma unroll
      for (int r = 0; r < 8; ++r)
        pl[(size_t)(mt * 16 + r) * N] = vals[r];   // C layout: row = r + 8*hi
    }
    if (lhalf == 0) {
      #pragma unroll
      for (int r = 0; r < 8; ++r) {
        unsigned long long key = ((unsigned long long)keyhi[r] << 32) |
                                 (unsigned long long)(0xFFFFFFFFu - nwin[r]);
        atomicMax(&amax64[aidx + mt * 16 + r], key);
      }
    }
  }
}

// ---------------------------------------------------------------------------
// Kernel 2: per-(o,n) column: max -> exp(p-max-dist/(2s^2)) -> incremental
// top-64 (only the winner's owner rescans its 32 private slots) ->
// normalize -> values-gather matvec + q_out*mask_mem.
// ---------------------------------------------------------------------------
__global__ __launch_bounds__(256)
void k_topk_out(const float* __restrict__ p,
                const unsigned long long* __restrict__ amax64,
                const float* __restrict__ values, const float* __restrict__ masks,
                const float* __restrict__ qout, float* __restrict__ out,
                const int* __restrict__ wptr, int N) {
  __shared__ float s[BANK];          // 32 KB column buffer (thread-private slots)
  __shared__ float rv[16];           // double-buffered wave results
  __shared__ int   ri[16];
  __shared__ float tv[TOPK];
  __shared__ int   ti[TOPK];

  const int n = blockIdx.x, o = blockIdx.y, t = threadIdx.x;
  const int wave = t >> 5, lane = t & 31;
  const int w = wptr[0];
  const float inv2s2 = 1.0f / (2.0f * KM_SIG * KM_SIG);

  // Load column + column max (stabilizer; constant cancels after renorm)
  float lmax = -3.4e38f;
  #pragma unroll
  for (int i = 0; i < BANK / 256; ++i) {
    int k = t + i * 256;
    float v = p[((size_t)o * BANK + k) * N + n];
    s[k] = v;
    lmax = fmaxf(lmax, v);
  }
  #pragma unroll
  for (int off = 1; off < 32; off <<= 1) lmax = fmaxf(lmax, __shfl_xor(lmax, off));
  if (lane == 0) rv[wave] = lmax;
  __syncthreads();
  float cmax = rv[0];
  #pragma unroll
  for (int i = 1; i < 8; ++i) cmax = fmaxf(cmax, rv[i]);
  __syncthreads();                   // rv free for reuse below

  // Fused exp(score - max - dist/(2 sigma^2)); track per-thread best
  const float yn = (float)(n / w), xn = (float)(n % w);
  float bv = -1.0f;
  int   bk = 0x7fffffff;
  #pragma unroll
  for (int i = 0; i < BANK / 256; ++i) {
    int k = t + i * 256;
    int an = (int)(0xFFFFFFFFu - (unsigned)(amax64[o * BANK + k] & 0xFFFFFFFFull));
    float dy = yn - (float)(an / w);
    float dx = xn - (float)(an % w);
    float e = expf(s[k] - cmax - (dy * dy + dx * dx) * inv2s2);
    s[k] = e;
    if (e > bv) { bv = e; bk = k; }
  }

  // Incremental top-64: only the extracted winner's owner thread rescans
  float wsum = 0.0f;
  for (int j = 0; j < TOPK; ++j) {
    float rvv = bv; int rkk = bk;
    #pragma unroll
    for (int off = 1; off < 32; off <<= 1) {
      float ov = __shfl_xor(rvv, off);
      int   oi = __shfl_xor(rkk, off);
      if (ov > rvv || (ov == rvv && oi < rkk)) { rvv = ov; rkk = oi; }
    }
    const int buf = (j & 1) * 8;
    if (lane == 0) { rv[buf + wave] = rvv; ri[buf + wave] = rkk; }
    __syncthreads();
    float wv = rv[buf]; int wk = ri[buf];
    #pragma unroll
    for (int i = 1; i < 8; ++i) {
      if (rv[buf + i] > wv || (rv[buf + i] == wv && ri[buf + i] < wk)) {
        wv = rv[buf + i]; wk = ri[buf + i];
      }
    }
    if (t == 0) { tv[j] = wv; ti[j] = wk; }
    wsum += wv;                      // identical in every thread
    if ((wk & 255) == t) {           // owner: tombstone + private rescan
      s[wk] = -1.0f;
      bv = -1.0f; bk = 0x7fffffff;
      #pragma unroll
      for (int i = 0; i < BANK / 256; ++i) {
        int k = t + i * 256;
        float v = s[k];
        if (v > bv) { bv = v; bk = k; }
      }
    }
  }
  __syncthreads();

  // Pre-scale weights once
  const float inv = 1.0f / wsum;
  if (t < TOPK) tv[t] *= inv;
  __syncthreads();

  // mask_mem scalar (broadcast loads; L2-resident)
  float msum = 0.0f;
  #pragma unroll
  for (int j = 0; j < TOPK; ++j) msum += tv[j] * masks[o * BANK + ti[j]];

  // Sparse matvec over 64 value-columns (values fit in 192MB L2)
  for (int v = t; v < DVAL; v += 256) {
    float acc = 0.0f;
    #pragma unroll
    for (int j = 0; j < TOPK; ++j)
      acc += tv[j] * values[((size_t)o * DVAL + v) * BANK + ti[j]];
    out[((size_t)o * 2 * DVAL + v) * N + n] = acc;                       // mem
    out[((size_t)o * 2 * DVAL + DVAL + v) * N + n] =
        qout[(size_t)v * N + n] * msum;                                  // q_out*mask_mem
  }
}

// ---------------------------------------------------------------------------
extern "C" void kernel_launch(void* const* d_in, const int* in_sizes, int n_in,
                              void* d_out, int out_size, void* d_ws, size_t ws_size,
                              hipStream_t stream) {
  const float* keys   = (const float*)d_in[0]; // [3,128,8192]
  const float* values = (const float*)d_in[1]; // [3,512,8192]
  const float* masks  = (const float*)d_in[2]; // [3,1,8192]
  const float* qin    = (const float*)d_in[3]; // [1,128,N]
  const float* qout   = (const float*)d_in[4]; // [1,512,N]
  const int*   wptr   = (const int*)d_in[6];   // w
  float* out = (float*)d_out;                  // [1,3,1024,N]

  const int N = in_sizes[3] / DKEY;            // 1620
  float* p = (float*)d_ws;                                         // O*BANK*N f32
  unsigned long long* amax64 =
      (unsigned long long*)(p + (size_t)O_N * BANK * N);           // O*BANK u64

  const int rows = O_N * BANK;
  k_init<<<dim3((rows + 255) / 256), 256, 0, stream>>>(amax64, rows);

  const int nblk = (N + 127) / 128;
  k_scores<<<dim3(nblk, BANK / 128, O_N), 256, 0, stream>>>(keys, qin, p, amax64, N);

  k_topk_out<<<dim3(N, O_N), 256, 0, stream>>>(p, amax64, values, masks, qout,
                                               out, wptr, N);
}
